// CalderaLinear_4166118277881
// MI455X (gfx1250) — matrix operand order repaired and checked by reading, not compiled, not run on hardware
//
#include <hip/hip_runtime.h>
#include <hip/hip_bf16.h>

typedef __attribute__((ext_vector_type(16))) _Float16 v16h;
typedef __attribute__((ext_vector_type(8)))  _Float16 v8h;
typedef __attribute__((ext_vector_type(8)))  float    v8f;
typedef __attribute__((ext_vector_type(4)))  float    v4f;
typedef __attribute__((ext_vector_type(4)))  int      v4i;

#define GROUP   128
#define TILE_K  32
#define LDW     40   // halfs per LDS row: 80 bytes, multiple of 16B for b128 LDS reads

// ---- WMMA fragment loaders (CDNA5 16-bit layouts, wave32) ----------------

// A matrix 16x32 (MxK): lane L -> row M = L&15; halfs = K {kk..kk+7, kk+16..kk+23}, kk = 8*(L>>4)
__device__ __forceinline__ v16h load_frag_a(const _Float16* lds, int rowBase, int lane) {
    const int row = rowBase + (lane & 15);
    const int kk  = (lane >> 4) * 8;
    v8h lo = *(const v8h*)(lds + row * LDW + kk);
    v8h hi = *(const v8h*)(lds + row * LDW + kk + 16);
    v16h a;
#pragma unroll
    for (int i = 0; i < 8; ++i) { a[i] = lo[i]; a[i + 8] = hi[i]; }
    return a;
}

// B matrix 32x16 (KxN): lane L -> col N = L&15; halfs = K {kk..kk+7, kk+16..kk+23}, kk = 8*(L>>4)
// Weight tile stored K-major per output row, so these are two 16B runs.
__device__ __forceinline__ v16h load_frag_b(const _Float16* lds, int rowBase, int lane) {
    const int row = rowBase + (lane & 15);
    const int kk  = (lane >> 4) * 8;
    v8h lo = *(const v8h*)(lds + row * LDW + kk);
    v8h hi = *(const v8h*)(lds + row * LDW + kk + 16);
    v16h b;
#pragma unroll
    for (int i = 0; i < 8; ++i) { b[i] = lo[i]; b[i + 8] = hi[i]; }
    return b;
}

// ---- Tile stagers ---------------------------------------------------------

// fp32 -> f16 activation tile: 128 rows x 32 cols, 256 threads, 16 elems/thread
__device__ __forceinline__ void stage_x_tile(_Float16* dst_lds, const float* __restrict__ src,
                                             int srcStride, int row0, int k0, int tid) {
    const int r  = tid >> 1;
    const int c0 = (tid & 1) * 16;
    const float* s = src + (size_t)(row0 + r) * srcStride + k0 + c0;
    _Float16* d = dst_lds + r * LDW + c0;
#pragma unroll
    for (int i = 0; i < 16; i += 4) {
        v4f v = *(const v4f*)(s + i);
        d[i + 0] = (_Float16)v[0];
        d[i + 1] = (_Float16)v[1];
        d[i + 2] = (_Float16)v[2];
        d[i + 3] = (_Float16)v[3];
    }
}

// int-code -> dequant f16 weight tile: 128 rows x 32 cols
__device__ __forceinline__ void stage_w_tile(_Float16* dst_lds, const int* __restrict__ vals,
                                             const float* __restrict__ scales,
                                             int K, int row0, int k0, int tid) {
    const int r  = tid >> 1;
    const int c0 = (tid & 1) * 16;
    const int orow = row0 + r;
    const float sc = scales[(size_t)orow * (K / GROUP) + (k0 + c0) / GROUP];
    const int* s = vals + (size_t)orow * K + k0 + c0;
    _Float16* d = dst_lds + r * LDW + c0;
#pragma unroll
    for (int i = 0; i < 16; i += 4) {
        v4i v = *(const v4i*)(s + i);
        d[i + 0] = (_Float16)((float)v[0] * sc);
        d[i + 1] = (_Float16)((float)v[1] * sc);
        d[i + 2] = (_Float16)((float)v[2] * sc);
        d[i + 3] = (_Float16)((float)v[3] * sc);
    }
}

// f16 activation tile (xr) straight copy: 128 rows x 32 cols
__device__ __forceinline__ void stage_h_tile(_Float16* dst_lds, const _Float16* __restrict__ src,
                                             int srcStride, int row0, int k0, int tid) {
    const int r  = tid >> 1;
    const int c0 = (tid & 1) * 16;
    const _Float16* s = src + (size_t)(row0 + r) * srcStride + k0 + c0;
    _Float16* d = dst_lds + r * LDW + c0;
    *(v8h*)(d)     = *(const v8h*)(s);
    *(v8h*)(d + 8) = *(const v8h*)(s + 8);
}

// ---- Kernel A: xr[T x RANK] (f16) = x[T x K] @ dequant(R)[RANK x K]^T -----

__global__ __launch_bounds__(256)
void caldera_xr_kernel(const float* __restrict__ x,
                       const int* __restrict__ r_values,
                       const float* __restrict__ r_scales,
                       _Float16* __restrict__ xr,
                       int T, int K, int RANK) {
    __shared__ _Float16 xs[128 * LDW];
    __shared__ _Float16 wsh[128 * LDW];
    const int tid  = threadIdx.x;
    const int lane = tid & 31;
    const int wave = tid >> 5;
    const int wave_m = wave & 3;   // 4 waves over M (4 x 32 = 128)
    const int wave_n = wave >> 2;  // 2 waves over N (2 x 64 = 128)
    const int mBlk = blockIdx.y * 128;
    const int nBlk = blockIdx.x * 128;

    v8f acc[2][4];
#pragma unroll
    for (int mi = 0; mi < 2; ++mi)
#pragma unroll
        for (int ni = 0; ni < 4; ++ni) acc[mi][ni] = (v8f)0.f;

    for (int k0 = 0; k0 < K; k0 += TILE_K) {
        stage_x_tile(xs,  x,        K, mBlk, k0, tid);
        stage_w_tile(wsh, r_values, r_scales, K, nBlk, k0, tid);
        __syncthreads();

        v16h a[2], b[4];
#pragma unroll
        for (int mi = 0; mi < 2; ++mi) a[mi] = load_frag_a(xs,  32 * wave_m + 16 * mi, lane);
#pragma unroll
        for (int ni = 0; ni < 4; ++ni) b[ni] = load_frag_b(wsh, 64 * wave_n + 16 * ni, lane);
#pragma unroll
        for (int mi = 0; mi < 2; ++mi)
#pragma unroll
            for (int ni = 0; ni < 4; ++ni)
                acc[mi][ni] = __builtin_amdgcn_wmma_f32_16x16x32_f16(
                    false, a[mi], false, b[ni], (short)0, acc[mi][ni], false, false);
        __syncthreads();
    }

#pragma unroll
    for (int mi = 0; mi < 2; ++mi) {
        const int tbase = mBlk + 32 * wave_m + 16 * mi + 8 * (lane >> 4);
#pragma unroll
        for (int ni = 0; ni < 4; ++ni) {
            const int o = nBlk + 64 * wave_n + 16 * ni + (lane & 15);
#pragma unroll
            for (int j = 0; j < 8; ++j)
                xr[(size_t)(tbase + j) * RANK + o] = (_Float16)acc[mi][ni][j];
        }
    }
}

// ---- Kernel B: out = x @ dequant(Q)^T + xr @ dequant(L)^T + bias ----------

__global__ __launch_bounds__(256)
void caldera_main_kernel(const float* __restrict__ x,
                         const int* __restrict__ q_values,
                         const float* __restrict__ q_scales,
                         const _Float16* __restrict__ xr,
                         const int* __restrict__ l_values,
                         const float* __restrict__ l_scales,
                         const float* __restrict__ bias,
                         float* __restrict__ out,
                         int T, int K, int OUT, int RANK) {
    __shared__ _Float16 xs[128 * LDW];
    __shared__ _Float16 wsh[128 * LDW];
    const int tid  = threadIdx.x;
    const int lane = tid & 31;
    const int wave = tid >> 5;
    const int wave_m = wave & 3;
    const int wave_n = wave >> 2;
    const int mBlk = blockIdx.y * 128;
    const int nBlk = blockIdx.x * 128;

    v8f acc[2][4];
#pragma unroll
    for (int mi = 0; mi < 2; ++mi)
#pragma unroll
        for (int ni = 0; ni < 4; ++ni) acc[mi][ni] = (v8f)0.f;

    // Phase 1: base GEMM over K = d_in
    for (int k0 = 0; k0 < K; k0 += TILE_K) {
        stage_x_tile(xs,  x,        K, mBlk, k0, tid);
        stage_w_tile(wsh, q_values, q_scales, K, nBlk, k0, tid);
        __syncthreads();

        v16h a[2], b[4];
#pragma unroll
        for (int mi = 0; mi < 2; ++mi) a[mi] = load_frag_a(xs,  32 * wave_m + 16 * mi, lane);
#pragma unroll
        for (int ni = 0; ni < 4; ++ni) b[ni] = load_frag_b(wsh, 64 * wave_n + 16 * ni, lane);
#pragma unroll
        for (int mi = 0; mi < 2; ++mi)
#pragma unroll
            for (int ni = 0; ni < 4; ++ni)
                acc[mi][ni] = __builtin_amdgcn_wmma_f32_16x16x32_f16(
                    false, a[mi], false, b[ni], (short)0, acc[mi][ni], false, false);
        __syncthreads();
    }

    // Phase 2: low-rank correction over K = rank, same accumulators
    for (int k0 = 0; k0 < RANK; k0 += TILE_K) {
        stage_h_tile(xs,  xr,       RANK, mBlk, k0, tid);
        stage_w_tile(wsh, l_values, l_scales, RANK, nBlk, k0, tid);
        __syncthreads();

        v16h a[2], b[4];
#pragma unroll
        for (int mi = 0; mi < 2; ++mi) a[mi] = load_frag_a(xs,  32 * wave_m + 16 * mi, lane);
#pragma unroll
        for (int ni = 0; ni < 4; ++ni) b[ni] = load_frag_b(wsh, 64 * wave_n + 16 * ni, lane);
#pragma unroll
        for (int mi = 0; mi < 2; ++mi)
#pragma unroll
            for (int ni = 0; ni < 4; ++ni)
                acc[mi][ni] = __builtin_amdgcn_wmma_f32_16x16x32_f16(
                    false, a[mi], false, b[ni], (short)0, acc[mi][ni], false, false);
        __syncthreads();
    }

    // Epilogue: + bias, fp32 store
#pragma unroll
    for (int mi = 0; mi < 2; ++mi) {
        const int tbase = mBlk + 32 * wave_m + 16 * mi + 8 * (lane >> 4);
#pragma unroll
        for (int ni = 0; ni < 4; ++ni) {
            const int o = nBlk + 64 * wave_n + 16 * ni + (lane & 15);
            const float bv = bias[o];
#pragma unroll
            for (int j = 0; j < 8; ++j)
                out[(size_t)(tbase + j) * OUT + o] = acc[mi][ni][j] + bv;
        }
    }
}

// ---- Launch ---------------------------------------------------------------

extern "C" void kernel_launch(void* const* d_in, const int* in_sizes, int n_in,
                              void* d_out, int out_size, void* d_ws, size_t ws_size,
                              hipStream_t stream) {
    const float* x        = (const float*)d_in[0];
    const int*   q_values = (const int*)  d_in[1];
    const float* q_scales = (const float*)d_in[2];
    const int*   l_values = (const int*)  d_in[3];
    const float* l_scales = (const float*)d_in[4];
    const int*   r_values = (const int*)  d_in[5];
    const float* r_scales = (const float*)d_in[6];
    const float* bias     = (const float*)d_in[7];
    float*       out      = (float*)d_out;

    const int OUT  = in_sizes[7];              // d_out from bias
    const int K    = in_sizes[1] / OUT;        // d_in  from q_values
    const int RANK = in_sizes[3] / OUT;        // rank  from l_values
    const int T    = in_sizes[0] / K;          // tokens from x

    _Float16* xr = (_Float16*)d_ws;            // T x RANK f16 scratch (4 MB)

    dim3 blk(256, 1, 1);
    dim3 gridA(RANK / 128, T / 128, 1);
    caldera_xr_kernel<<<gridA, blk, 0, stream>>>(x, r_values, r_scales, xr, T, K, RANK);

    dim3 gridB(OUT / 128, T / 128, 1);
    caldera_main_kernel<<<gridB, blk, 0, stream>>>(x, q_values, q_scales, xr,
                                                   l_values, l_scales, bias, out,
                                                   T, K, OUT, RANK);
}
